// YoloV9Trainer_44152263803091
// MI455X (gfx1250) — matrix operand order repaired and checked by reading, not compiled, not run on hardware
//
#include <hip/hip_runtime.h>
#include <math.h>
#include <float.h>

#define BATCH 2
#define TT    100
#define CC    80
#define EE    32
#define AA    8400
#define MHW   64
#define HW    4096
#define OUTC  8363
#define EPSF  1e-9f
#define FOUR_OVER_PI2 0.405284734569351f

typedef __attribute__((ext_vector_type(16))) _Float16 v16h;
typedef __attribute__((ext_vector_type(8)))  float    v8f;

// fast sigmoid: v_exp_f32 + v_rcp_f32 (1 ulp each) — plenty for fp32 outputs
__device__ __forceinline__ float fast_sigmoid(float x) {
    return __builtin_amdgcn_rcpf(1.f + __expf(-x));
}

// ---------------------------------------------------------------------------
// Kernel 1: per (b,t,a): grid_mask * clip(ciou,0,1)^6 * sqrt(sigmoid(cls))
// ---------------------------------------------------------------------------
__global__ void k_tmat(const float* __restrict__ cls_logits,
                       const float* __restrict__ pred_boxes,
                       const int*   __restrict__ target_cls,
                       const float* __restrict__ target_bbox,
                       const float* __restrict__ anchors,
                       float* __restrict__ tmat,
                       float* __restrict__ iouA)
{
    int idx = blockIdx.x * blockDim.x + threadIdx.x;
    const int total = BATCH * TT * AA;
    if (idx >= total) return;
    int a  = idx % AA;
    int bt = idx / AA;
    int t  = bt % TT;
    int b  = bt / TT;

    const float* g = target_bbox + (size_t)(b * TT + t) * 4;
    const float* p = pred_boxes  + (size_t)(b * AA + a) * 4;
    float gx1 = g[0], gy1 = g[1], gx2 = g[2], gy2 = g[3];
    float px1 = p[0], py1 = p[1], px2 = p[2], py2 = p[3];
    float ax = anchors[a * 2 + 0], ay = anchors[a * 2 + 1];

    float dmin = fminf(fminf(ax - gx1, ay - gy1), fminf(gx2 - ax, gy2 - ay));
    float gm   = (dmin > 0.f) ? 1.f : 0.f;

    float ix1 = fmaxf(gx1, px1), iy1 = fmaxf(gy1, py1);
    float ix2 = fminf(gx2, px2), iy2 = fminf(gy2, py2);
    float inter = fmaxf(ix2 - ix1, 0.f) * fmaxf(iy2 - iy1, 0.f);
    float wg = gx2 - gx1, hg = gy2 - gy1;
    float wp = px2 - px1, hp = py2 - py1;
    float uni = wg * hg + wp * hp - inter + EPSF;
    float iou = inter / uni;
    float cw = fmaxf(gx2, px2) - fminf(gx1, px1);
    float ch = fmaxf(gy2, py2) - fminf(gy1, py1);
    float c2 = cw * cw + ch * ch + EPSF;
    float dx = gx1 + gx2 - px1 - px2;
    float dy = gy1 + gy2 - py1 - py2;
    float rho2 = (dx * dx + dy * dy) * 0.25f;
    float da = atanf(wg / (hg + EPSF)) - atanf(wp / (hp + EPSF));
    float v  = FOUR_OVER_PI2 * da * da;
    float alpha = v / (1.f - iou + v + EPSF);
    float ciou  = iou - rho2 / c2 - alpha * v;
    float iou_c = fminf(fmaxf(ciou, 0.f), 1.f);

    int   tc  = target_cls[b * TT + t];
    float lg  = cls_logits[(size_t)(b * AA + a) * CC + tc];
    float sig = fast_sigmoid(lg);

    float i2 = iou_c * iou_c;
    tmat[idx] = gm * (i2 * i2 * i2) * sqrtf(sig);
    iouA[idx] = iou_c;
}

// ---------------------------------------------------------------------------
// Kernel 2: per (b,t) row: max(tmat), max(iou), 10th-largest (with ties)
// ---------------------------------------------------------------------------
__global__ void k_rowstats(const float* __restrict__ tmat,
                           const float* __restrict__ iouA,
                           float* __restrict__ kth,
                           float* __restrict__ maxt,
                           float* __restrict__ maxiou)
{
    int row = blockIdx.x; // 0 .. BATCH*TT-1
    const float* tr = tmat + (size_t)row * AA;
    const float* ir = iouA + (size_t)row * AA;
    __shared__ float sf[256];
    __shared__ int   si[256];
    int tid = threadIdx.x;

    float m1 = -FLT_MAX, m2 = -FLT_MAX;
    for (int a = tid; a < AA; a += 256) {
        m1 = fmaxf(m1, tr[a]);
        m2 = fmaxf(m2, ir[a]);
    }
    sf[tid] = m1; __syncthreads();
    for (int s = 128; s > 0; s >>= 1) { if (tid < s) sf[tid] = fmaxf(sf[tid], sf[tid + s]); __syncthreads(); }
    float mt = sf[0]; __syncthreads();
    sf[tid] = m2; __syncthreads();
    for (int s = 128; s > 0; s >>= 1) { if (tid < s) sf[tid] = fmaxf(sf[tid], sf[tid + s]); __syncthreads(); }
    float mi = sf[0]; __syncthreads();

    // kth largest counted with multiplicity (TOPK = 10)
    float thresh = FLT_MAX;
    float kv = 0.f;
    int total = 0;
    for (int it = 0; it < 10; ++it) {
        float lm = -FLT_MAX;
        for (int a = tid; a < AA; a += 256) {
            float vv = tr[a];
            if (vv < thresh) lm = fmaxf(lm, vv);
        }
        sf[tid] = lm; __syncthreads();
        for (int s = 128; s > 0; s >>= 1) { if (tid < s) sf[tid] = fmaxf(sf[tid], sf[tid + s]); __syncthreads(); }
        float mx = sf[0]; __syncthreads();
        int lc = 0;
        for (int a = tid; a < AA; a += 256) if (tr[a] == mx) lc++;
        si[tid] = lc; __syncthreads();
        for (int s = 128; s > 0; s >>= 1) { if (tid < s) si[tid] += si[tid + s]; __syncthreads(); }
        int c = si[0]; __syncthreads();
        total += c;
        kv = mx;
        if (total >= 10 || mx <= 0.f) break;  // uniform across block
        thresh = mx;
    }
    if (tid == 0) { kth[row] = kv; maxt[row] = mt; maxiou[row] = mi; }
}

// ---------------------------------------------------------------------------
// Kernel 3: per (b,a): assignment + output cols 0..168, 4265..4266
// ---------------------------------------------------------------------------
__global__ void k_assign(const float* __restrict__ cls_logits,
                         const float* __restrict__ pred_boxes,
                         const int*   __restrict__ target_cls,
                         const float* __restrict__ target_bbox,
                         const float* __restrict__ scalers,
                         const float* __restrict__ tmat,
                         const float* __restrict__ kth,
                         const float* __restrict__ maxt,
                         const float* __restrict__ maxiou,
                         int*   __restrict__ uidx,
                         float* __restrict__ validf,
                         float* __restrict__ bounds,
                         float* __restrict__ out)
{
    int idx = blockIdx.x * blockDim.x + threadIdx.x;
    if (idx >= BATCH * AA) return;
    int b = idx / AA, a = idx % AA;

    float best = -1.f, ssum = 0.f;
    int u = 0;
    for (int t = 0; t < TT; ++t) {
        float v  = tmat[(size_t)(b * TT + t) * AA + a];
        float th = kth[b * TT + t];
        float tk = (v >= th && v > 0.f) ? v : 0.f;
        ssum += tk;
        if (tk > best) { best = tk; u = t; }
    }
    float vf = (ssum > 0.f) ? 1.f : 0.f;

    const float* tb = target_bbox + (size_t)(b * TT + u) * 4;
    int   tcu = target_cls[b * TT + u];
    float ng  = tmat[(size_t)(b * TT + u) * AA + a] / (maxt[b * TT + u] + EPSF) * maxiou[b * TT + u];
    float inv = 1.f / scalers[a];

    float* o = out + (size_t)idx * OUTC;
    const float* p = pred_boxes + (size_t)idx * 4;
    o[0] = p[0] * inv; o[1] = p[1] * inv; o[2] = p[2] * inv; o[3] = p[3] * inv;
    const float* cl = cls_logits + (size_t)idx * CC;
    for (int j = 0; j < CC; ++j) o[4 + j] = cl[j];
    float bx1 = tb[0], by1 = tb[1], bx2 = tb[2], by2 = tb[3];
    o[84] = bx1 * inv; o[85] = by1 * inv; o[86] = bx2 * inv; o[87] = by2 * inv;
    o[88] = vf;
    float ac = ng * vf;
    for (int j = 0; j < CC; ++j) o[89 + j] = (j == tcu) ? ac : 0.f;
    o[4265] = (bx2 - bx1) * (by2 - by1) * (1.f / (640.f * 640.f));
    o[4266] = vf;

    uidx[idx]   = u;
    validf[idx] = vf;
    bounds[idx * 4 + 0] = bx1 * 0.1f;   // 64/640 scale to mask grid
    bounds[idx * 4 + 1] = by1 * 0.1f;
    bounds[idx * 4 + 2] = bx2 * 0.1f;
    bounds[idx * 4 + 3] = by2 * 0.1f;
}

// ---------------------------------------------------------------------------
// Kernel 4: WMMA GEMM (mask_embs x protos^T), fused sigmoid + box-mask +
// target_masks gather. One v_wmma_f32_16x16x32_f16 per 16x16 tile (K=32).
// Tiles whose 16 anchor rows are all invalid take a store-only zero path.
// ---------------------------------------------------------------------------
__global__ void __launch_bounds__(256)
k_seg(const float* __restrict__ mask_embs,
      const float* __restrict__ protos,
      const float* __restrict__ target_masks,
      const int*   __restrict__ uidx,
      const float* __restrict__ validf,
      const float* __restrict__ bounds,
      float* __restrict__ out)
{
    int b     = blockIdx.y;
    int mtile = blockIdx.x >> 5;    // 0..524  (8400 / 16)
    int ngrp  = blockIdx.x & 31;    // 0..31   (4096 / 128)
    int wave  = threadIdx.x >> 5;   // 0..7
    int lane  = threadIdx.x & 31;
    int m0 = mtile * 16;
    int n0 = (ngrp * 8 + wave) * 16;
    int  lh = lane & 15;
    bool hi = lane >= 16;
    int col = n0 + lh;

    // Wave-uniform validity check: lane i holds valid flag of anchor row m0+i.
    float rowvalid = validf[b * AA + m0 + lh];
    bool tile_live = __any(rowvalid > 0.f);

    if (!tile_live) {
        // All 16 rows unassigned: seg_pred and align_seg are exactly zero.
        #pragma unroll
        for (int r = 0; r < 8; ++r) {
            size_t base = (size_t)(b * AA + m0 + r + (hi ? 8 : 0)) * OUTC;
            out[base + 169  + col] = 0.f;
            out[base + 4267 + col] = 0.f;
        }
        return;
    }

    // A (16x32 f16): lane-low holds K 0..7 & 16..23 of its row; lane-high 8..15 & 24..31
    const float* pA = mask_embs + (size_t)(b * AA + m0 + lh) * EE + (hi ? 8 : 0);
    float4 fa0 = *(const float4*)(pA);
    float4 fa1 = *(const float4*)(pA + 4);
    float4 fa2 = *(const float4*)(pA + 16);
    float4 fa3 = *(const float4*)(pA + 20);
    // B (32x16 f16) = protos^T: lane-low holds K 0..7 & 16..23 of its column; lane-high 8..15 & 24..31
    const float* pB = protos + (size_t)(b * HW + n0 + lh) * EE + (hi ? 8 : 0);
    float4 fb0 = *(const float4*)(pB);
    float4 fb1 = *(const float4*)(pB + 4);
    float4 fb2 = *(const float4*)(pB + 16);
    float4 fb3 = *(const float4*)(pB + 20);

    v16h av, bv;
    av[0]  = (_Float16)fa0.x; av[1]  = (_Float16)fa0.y; av[2]  = (_Float16)fa0.z; av[3]  = (_Float16)fa0.w;
    av[4]  = (_Float16)fa1.x; av[5]  = (_Float16)fa1.y; av[6]  = (_Float16)fa1.z; av[7]  = (_Float16)fa1.w;
    av[8]  = (_Float16)fa2.x; av[9]  = (_Float16)fa2.y; av[10] = (_Float16)fa2.z; av[11] = (_Float16)fa2.w;
    av[12] = (_Float16)fa3.x; av[13] = (_Float16)fa3.y; av[14] = (_Float16)fa3.z; av[15] = (_Float16)fa3.w;
    bv[0]  = (_Float16)fb0.x; bv[1]  = (_Float16)fb0.y; bv[2]  = (_Float16)fb0.z; bv[3]  = (_Float16)fb0.w;
    bv[4]  = (_Float16)fb1.x; bv[5]  = (_Float16)fb1.y; bv[6]  = (_Float16)fb1.z; bv[7]  = (_Float16)fb1.w;
    bv[8]  = (_Float16)fb2.x; bv[9]  = (_Float16)fb2.y; bv[10] = (_Float16)fb2.z; bv[11] = (_Float16)fb2.w;
    bv[12] = (_Float16)fb3.x; bv[13] = (_Float16)fb3.y; bv[14] = (_Float16)fb3.z; bv[15] = (_Float16)fb3.w;

    v8f c = {};
    c = __builtin_amdgcn_wmma_f32_16x16x32_f16(
        /*neg_a=*/false, av, /*neg_b=*/false, bv,
        /*c_mod=*/(short)0, c, /*reuse_a=*/false, /*reuse_b=*/false);

    int h = col >> 6, w = col & 63;
    float hf = (float)h, wf = (float)w;

    #pragma unroll
    for (int r = 0; r < 8; ++r) {
        int m  = m0 + r + (hi ? 8 : 0);
        int ga = b * AA + m;
        float vf = validf[ga];
        float x1 = bounds[ga * 4 + 0], y1 = bounds[ga * 4 + 1];
        float x2 = bounds[ga * 4 + 2], y2 = bounds[ga * 4 + 3];
        bool inb = (hf >= y1) && (hf < y2) && (wf >= x1) && (wf < x2);
        bool on  = (vf > 0.f) && inb;
        float sp = 0.f, tm = 0.f;
        if (on) {
            sp = fast_sigmoid(c[r]);
            int u = uidx[ga];
            tm = target_masks[((size_t)(b * MHW + h) * MHW + w) * TT + u];
        }
        size_t base = (size_t)ga * OUTC;
        out[base + 169  + col] = sp;
        out[base + 4267 + col] = tm;
    }
}

// ---------------------------------------------------------------------------
extern "C" void kernel_launch(void* const* d_in, const int* in_sizes, int n_in,
                              void* d_out, int out_size, void* d_ws, size_t ws_size,
                              hipStream_t stream) {
    (void)in_sizes; (void)n_in; (void)out_size; (void)ws_size;
    const float* cls_logits   = (const float*)d_in[0];
    const float* pred_boxes   = (const float*)d_in[1];
    const float* mask_embs    = (const float*)d_in[2];
    const float* protos       = (const float*)d_in[3];
    const int*   target_cls   = (const int*)  d_in[4];
    const float* target_bbox  = (const float*)d_in[5];
    const float* target_masks = (const float*)d_in[6];
    const float* anchors      = (const float*)d_in[7];
    const float* scalers      = (const float*)d_in[8];
    float* out = (float*)d_out;

    float* ws      = (float*)d_ws;
    float* tmat    = ws;                                   // B*T*A
    float* iouA    = tmat + (size_t)BATCH * TT * AA;       // B*T*A
    float* kth     = iouA + (size_t)BATCH * TT * AA;       // B*T
    float* maxt    = kth + BATCH * TT;                     // B*T
    float* maxiou  = maxt + BATCH * TT;                    // B*T
    float* validfA = maxiou + BATCH * TT;                  // B*A
    float* boundsA = validfA + (size_t)BATCH * AA;         // B*A*4
    int*   uidxA   = (int*)(boundsA + (size_t)BATCH * AA * 4); // B*A

    int totalA = BATCH * TT * AA;
    k_tmat<<<(totalA + 255) / 256, 256, 0, stream>>>(
        cls_logits, pred_boxes, target_cls, target_bbox, anchors, tmat, iouA);
    k_rowstats<<<BATCH * TT, 256, 0, stream>>>(tmat, iouA, kth, maxt, maxiou);
    k_assign<<<(BATCH * AA + 255) / 256, 256, 0, stream>>>(
        cls_logits, pred_boxes, target_cls, target_bbox, scalers,
        tmat, kth, maxt, maxiou, uidxA, validfA, boundsA, out);
    dim3 g(525 * 32, BATCH);
    k_seg<<<g, 256, 0, stream>>>(mask_embs, protos, target_masks,
                                 uidxA, validfA, boundsA, out);
}
